// MMDLoss_15668040696108
// MI455X (gfx1250) — compile-verified, matrix operand order checked by reading, not run on hardware
//
#include <hip/hip_runtime.h>

typedef __attribute__((ext_vector_type(16))) __bf16 v16bf;
typedef __attribute__((ext_vector_type(8)))  __bf16 v8bf;
typedef __attribute__((ext_vector_type(8)))  float  v8f;
typedef __attribute__((ext_vector_type(4)))  int    v4i;

#define N_ROWS 4096
#define N_COLS 512

// Macro-tile geometry: 128x128 C-block per workgroup, K-step 32, double-buffered LDS.
#define BM 128
#define BN 128
#define BK 32
#define NSTEP (N_COLS / BK)          // 16
#define LS 40                         // LDS row stride in bf16 (80B = 20 banks: conflict-free)
#define BLOCKS_1D (N_ROWS / BM)       // 32
#define BLOCKS_PER_PAIR (BLOCKS_1D * BLOCKS_1D)  // 1024
// Wave grid: 2 (M) x 4 (N); each wave: 64x32 of C = 4x2 sub-tiles of 16x16.
#define WAVES_PER_BLOCK 8

// s_wait_asynccnt with literal immediate (ASYNCcnt <= n).
#define WAIT_ASYNC(lit) asm volatile("s_wait_asynccnt " lit ::: "memory")

// One 16-byte async global->LDS transfer (VGLOBAL GLOBAL_LOAD_ASYNC_TO_LDS_B128).
// ROCm clang declares the builtin with typed pointers: (global v4i*, lds v4i*, imm, imm).
__device__ __forceinline__ void async_copy_b128(const __bf16* g, __bf16* l) {
#if __has_builtin(__builtin_amdgcn_global_load_async_to_lds_b128)
  typedef __attribute__((address_space(1))) v4i g_v4i;
  typedef __attribute__((address_space(3))) v4i l_v4i;
  __builtin_amdgcn_global_load_async_to_lds_b128(
      (g_v4i*)(uintptr_t)g, (l_v4i*)(uintptr_t)l, 0, 0);
#else
  // LDS addresses are the low 32 bits of the generic pointer (ISA 10.2).
  unsigned lofs = (unsigned)(uintptr_t)l;
  asm volatile("global_load_async_to_lds_b128 %0, %1, off"
               :: "v"(lofs), "v"(g) : "memory");
#endif
}

// Assemble a 16x32-per-wave WMMA operand fragment from two 16B LDS halves
// (keeps 16B alignment so the LS=40 swizzle-free stride is legal).
__device__ __forceinline__ v16bf ld_frag(const __bf16* p) {
  v8bf lo = *(const v8bf*)p;
  v8bf hi = *(const v8bf*)(p + 8);
  return __builtin_shufflevector(lo, hi, 0, 1, 2, 3, 4, 5, 6, 7,
                                 8, 9, 10, 11, 12, 13, 14, 15);
}

// ---------------------------------------------------------------------------
// Prep: f32 -> bf16 + row squared-norms from the rounded bf16 values.
// ---------------------------------------------------------------------------
__global__ __launch_bounds__(256) void mmd_prep_kernel(
    const float* __restrict__ lbl, const float* __restrict__ pred,
    __bf16* __restrict__ lblb, __bf16* __restrict__ predb,
    float* __restrict__ nl, float* __restrict__ np) {
  const int row = blockIdx.x;
  const int mat = blockIdx.y;
  const float* src = mat ? pred : lbl;
  __bf16*      dst = mat ? predb : lblb;
  float*       nrm = mat ? np : nl;

  const float* r = src + (size_t)row * N_COLS;
  __bf16*      w = dst + (size_t)row * N_COLS;

  float s = 0.0f;
  for (int c = threadIdx.x; c < N_COLS; c += blockDim.x) {
    __bf16 hv = (__bf16)r[c];
    w[c] = hv;
    float v = (float)hv;
    s += v * v;
  }
  for (int off = 16; off; off >>= 1) s += __shfl_xor(s, off, 32);

  __shared__ float ls[WAVES_PER_BLOCK];
  const int lane = threadIdx.x & 31;
  const int wave = threadIdx.x >> 5;
  if (lane == 0) ls[wave] = s;
  __syncthreads();
  if (threadIdx.x == 0) {
    float t = 0.0f;
#pragma unroll
    for (int i = 0; i < WAVES_PER_BLOCK; ++i) t += ls[i];
    nrm[row] = t;
  }
}

__global__ void mmd_init_kernel(float* __restrict__ acc) { acc[0] = 0.0f; }

// ---------------------------------------------------------------------------
// Fused MMD kernel: 128x128 macro-tile per block, async-DMA double-buffered
// LDS staging, 8 v_wmma per K-step per wave, fused exp + reduction epilogue.
// grid = (1024, 3 pairs), block = 256 threads = 8 waves (2x4).
// ---------------------------------------------------------------------------
__global__ __launch_bounds__(256) void mmd_wmma_kernel(
    const __bf16* __restrict__ lblb, const __bf16* __restrict__ predb,
    const float* __restrict__ nl, const float* __restrict__ np,
    float* __restrict__ acc) {
  const int pair = blockIdx.y;
  const int tm = blockIdx.x >> 5;   // macro-tile row
  const int tn = blockIdx.x & 31;   // macro-tile col

  // Symmetric pairs (ll, pp): only upper triangle; off-diagonal counts twice.
  if (pair < 2 && tm > tn) return;

  const __bf16* X; const __bf16* Y; const float* nx; const float* ny; float coef;
  if (pair == 0)      { X = lblb;  Y = lblb;  nx = nl; ny = nl; coef = (tm == tn) ? 1.0f : 2.0f; }
  else if (pair == 1) { X = predb; Y = predb; nx = np; ny = np; coef = (tm == tn) ? 1.0f : 2.0f; }
  else                { X = lblb;  Y = predb; nx = nl; ny = np; coef = -2.0f; }

  __shared__ __bf16 As[2][BM * LS];
  __shared__ __bf16 Bs[2][BM * LS];

  const int tid     = threadIdx.x;
  const int lane    = tid & 31;
  const int wave    = tid >> 5;
  const int wm      = wave >> 2;    // 0..1  -> 64-row slab
  const int wn      = wave & 3;     // 0..3  -> 32-col slab
  const int halfsel = lane >> 4;    // lanes 16-31 carry K+16 / M+8 halves
  const int lmod    = lane & 15;

  const size_t rowA0 = (size_t)(tm * BM) * N_COLS;
  const size_t rowB0 = (size_t)(tn * BM) * N_COLS;

  // Issue one K-step's DMA: 128x32 bf16 slab each for A and B (4 b128/thread).
  auto issue = [&](int ks, int buf) {
    const int k0 = ks * BK;
#pragma unroll
    for (int h = 0; h < 2; ++h) {
      const int t   = tid + h * 256;
      const int row = t >> 2;             // 0..127
      const int c8  = (t & 3) * 8;        // 0,8,16,24 (bf16 elements)
      async_copy_b128(X + rowA0 + (size_t)row * N_COLS + k0 + c8,
                      &As[buf][row * LS + c8]);
      async_copy_b128(Y + rowB0 + (size_t)row * N_COLS + k0 + c8,
                      &Bs[buf][row * LS + c8]);
    }
  };

  v8f c[4][2] = {};

  issue(0, 0);
#pragma unroll 2
  for (int ks = 0; ks < NSTEP; ++ks) {
    const int buf = ks & 1;
    if (ks + 1 < NSTEP) {
      issue(ks + 1, buf ^ 1);
      WAIT_ASYNC("4");   // current buffer's 4 transfers done; next 4 in flight
    } else {
      WAIT_ASYNC("0");
    }
    __syncthreads();

    // A fragments: rows wm*64 + i*16 + lmod; B fragments: rows wn*32 + j*16 + lmod.
    v16bf af[4], bf[2];
#pragma unroll
    for (int i = 0; i < 4; ++i)
      af[i] = ld_frag(&As[buf][(wm * 64 + i * 16 + lmod) * LS + halfsel * 16]);
#pragma unroll
    for (int j = 0; j < 2; ++j)
      bf[j] = ld_frag(&Bs[buf][(wn * 32 + j * 16 + lmod) * LS + halfsel * 16]);

#pragma unroll
    for (int i = 0; i < 4; ++i)
#pragma unroll
      for (int j = 0; j < 2; ++j)
        c[i][j] = __builtin_amdgcn_wmma_f32_16x16x32_bf16(
            false, af[i], false, bf[j], (short)0, c[i][j], false, false);

    __syncthreads();
  }

  // Epilogue: C layout -> VGPR r, lane l holds (M = r + 8*(l>=16), N = l&15).
  const int Mbase = tm * BM + wm * 64;
  const int Nbase = tn * BM + wn * 32;
  const float yy0 = ny[Nbase + lmod];
  const float yy1 = ny[Nbase + 16 + lmod];
  float s = 0.0f;
#pragma unroll
  for (int i = 0; i < 4; ++i) {
#pragma unroll
    for (int r = 0; r < 8; ++r) {
      const float xxv = nx[Mbase + i * 16 + halfsel * 8 + r];
      s += __expf(c[i][0][r] - 0.5f * (xxv + yy0));
      s += __expf(c[i][1][r] - 0.5f * (xxv + yy1));
    }
  }

  for (int off = 16; off; off >>= 1) s += __shfl_xor(s, off, 32);
  __shared__ float lsum[WAVES_PER_BLOCK];
  if (lane == 0) lsum[wave] = s * coef;
  __syncthreads();
  if (tid == 0) {
    float t = 0.0f;
#pragma unroll
    for (int i = 0; i < WAVES_PER_BLOCK; ++i) t += lsum[i];
    atomicAdd(acc, t);
  }
}

__global__ void mmd_final_kernel(const float* __restrict__ acc,
                                 float* __restrict__ out) {
  out[0] = acc[0] * (1.0f / ((float)N_ROWS * (float)N_ROWS));
}

extern "C" void kernel_launch(void* const* d_in, const int* in_sizes, int n_in,
                              void* d_out, int out_size, void* d_ws, size_t ws_size,
                              hipStream_t stream) {
  (void)in_sizes; (void)n_in; (void)out_size; (void)ws_size;

  const float* lbl  = (const float*)d_in[0];
  const float* pred = (const float*)d_in[1];
  float* out = (float*)d_out;

  char* w = (char*)d_ws;
  const size_t mat_bytes = (size_t)N_ROWS * N_COLS * sizeof(__bf16);  // 4 MB
  __bf16* lblb  = (__bf16*)(w);
  __bf16* predb = (__bf16*)(w + mat_bytes);
  float*  nl    = (float*)(w + 2 * mat_bytes);
  float*  np    = (float*)(w + 2 * mat_bytes + N_ROWS * sizeof(float));
  float*  acc   = (float*)(w + 2 * mat_bytes + 2 * N_ROWS * sizeof(float));

  mmd_prep_kernel<<<dim3(N_ROWS, 2), 256, 0, stream>>>(lbl, pred, lblb, predb, nl, np);
  mmd_init_kernel<<<1, 1, 0, stream>>>(acc);
  mmd_wmma_kernel<<<dim3(BLOCKS_PER_PAIR, 3), 256, 0, stream>>>(lblb, predb, nl, np, acc);
  mmd_final_kernel<<<1, 1, 0, stream>>>(acc, out);
}
